// InducedKernelGPR_163208757698
// MI455X (gfx1250) — compile-verified, matrix operand order checked by reading, not run on hardware
//
#include <hip/hip_runtime.h>
#include <hip/hip_fp16.h>

typedef __attribute__((ext_vector_type(16))) _Float16 v16h;
typedef __attribute__((ext_vector_type(8)))  _Float16 v8h;
typedef __attribute__((ext_vector_type(8)))  float    v8f;
typedef int v4i_vs __attribute__((vector_size(16)));   // matches async-LDS builtin param

#define D_DIM 256            // descriptor dimension (reference: 256)
#define PTILE 16             // inducing-point rows per WMMA tile
#define WAVES_PER_BLOCK 8
#define NCOLS_PER_BLOCK 128  // 8 waves * 16 columns
#define LDS_A_STRIDE 264     // halfs; 256 + 8 pad to break bank alignment

#if __has_builtin(__builtin_amdgcn_global_load_async_to_lds_b128)
#define ASYNC_LDS 1
#endif

#define AS_GLOBAL __attribute__((address_space(1)))
#define AS_LOCAL  __attribute__((address_space(3)))

// Stage one 32B chunk (16 halfs) of the zs tile into LDS.
__device__ __forceinline__ void stage_chunk(const _Float16* src, _Float16* dst) {
#ifdef ASYNC_LDS
  __builtin_amdgcn_global_load_async_to_lds_b128(
      (AS_GLOBAL v4i_vs*)src, (AS_LOCAL v4i_vs*)dst, /*offset=*/0, /*cpol=*/0);
  __builtin_amdgcn_global_load_async_to_lds_b128(
      (AS_GLOBAL v4i_vs*)(src + 8), (AS_LOCAL v4i_vs*)(dst + 8), /*offset=*/0, /*cpol=*/0);
#else
  *(v8h*)dst       = *(const v8h*)src;
  *(v8h*)(dst + 8) = *(const v8h*)(src + 8);
#endif
}

template <int N>
__device__ __forceinline__ void wait_async() {
#ifdef ASYNC_LDS
#if __has_builtin(__builtin_amdgcn_s_wait_asynccnt)
  __builtin_amdgcn_s_wait_asynccnt(N);
#else
  asm volatile("s_wait_asynccnt %0" :: "i"(N) : "memory");
#endif
#endif
}

// ---------------------------------------------------------------------------
// Prep: scale one row by exp(-0.5*ls), emit f16 row + f32 sum of squares.
// ---------------------------------------------------------------------------
__global__ __launch_bounds__(256)
void gpr_prep(const float* __restrict__ in, const float* __restrict__ ls,
              _Float16* __restrict__ out_h, float* __restrict__ out_sq, int d) {
  __shared__ float red[WAVES_PER_BLOCK];
  const int row = blockIdx.x;
  const int t   = threadIdx.x;
  float s = 0.0f;
  if (t < d) {
    float scale = __expf(-0.5f * ls[t]);
    float v = in[(size_t)row * d + t] * scale;
    out_h[(size_t)row * d + t] = (_Float16)v;
    s = v * v;
  }
  #pragma unroll
  for (int off = 16; off > 0; off >>= 1) s += __shfl_down(s, off, 32);
  if ((t & 31) == 0) red[t >> 5] = s;
  __syncthreads();
  if (t == 0) {
    float tot = 0.0f;
    #pragma unroll
    for (int i = 0; i < WAVES_PER_BLOCK; ++i) tot += red[i];
    out_sq[row] = tot;
  }
}

// ---------------------------------------------------------------------------
// Main: fused  out[n] = sum_p alpha[p] * exp(-(z2[p] + x2[n] - 2*zs.xs))
// 8 waves / 128 N-columns per block; wave w owns columns nBase+16w..+15.
// B fragments (16 cols x 256 K of xs) pinned in VGPRs for the whole P loop;
// zs tiles double-buffered through LDS via async global->LDS loads.
// ---------------------------------------------------------------------------
__global__ __launch_bounds__(256)
void gpr_main(const _Float16* __restrict__ xs, const _Float16* __restrict__ zs,
              const float* __restrict__ x_sq, const float* __restrict__ z_sq,
              const float* __restrict__ alpha, float* __restrict__ out, int P) {
  __shared__ _Float16 ldsA[2][PTILE * LDS_A_STRIDE];
  __shared__ float    ldsZsq[PTILE];
  __shared__ float    ldsAlpha[PTILE];

  const int lane  = threadIdx.x & 31;
  const int wave  = threadIdx.x >> 5;
  const int hi    = lane >> 4;           // half-wave select
  const int koff  = hi * 8;              // K offset for this half-wave
  const int nBase = blockIdx.x * NCOLS_PER_BLOCK;
  const int col   = nBase + wave * 16 + (lane & 15);   // this lane's N column

  // Per-thread slot in the cooperative tile stage: 256 chunks of 16 halfs.
  const int sc_r  = threadIdx.x >> 4;          // tile row 0..15
  const int sc_cc = (threadIdx.x & 15) * 16;   // half offset within row

  // --- Preload B fragments: element e holds K = kbase + koff + (e<8 ? e : e+8).
  v16h bfrag[8];
  {
    const _Float16* xrow = xs + (size_t)col * D_DIM;
    #pragma unroll
    for (int kk = 0; kk < 8; ++kk) {
      const int kb = kk * 32 + koff;
      v8h b0 = *(const v8h*)(xrow + kb);
      v8h b1 = *(const v8h*)(xrow + kb + 16);
      #pragma unroll
      for (int i = 0; i < 8; ++i) { bfrag[kk][i] = b0[i]; bfrag[kk][i + 8] = b1[i]; }
    }
  }

  const float my_xsq = x_sq[col];
  const int   a_row  = lane & 15;        // M index of A rows this lane reads
  v8f acc = {};                          // per-(lane,r) alpha*exp accumulator
  const int ntiles = P / PTILE;

  // Prime the pipeline: tile 0 -> buffer 0.
  stage_chunk(zs + (size_t)sc_r * D_DIM + sc_cc,
              &ldsA[0][sc_r * LDS_A_STRIDE + sc_cc]);

  for (int i = 0; i < ntiles; ++i) {
    const int pt = i * PTILE;

    if (i + 1 < ntiles) {
      // Issue next tile into the other buffer, then wait only for tile i
      // (each thread keeps <=2 newer async ops in flight).
      stage_chunk(zs + (size_t)(pt + PTILE + sc_r) * D_DIM + sc_cc,
                  &ldsA[(i + 1) & 1][sc_r * LDS_A_STRIDE + sc_cc]);
      if (i + 2 < ntiles)  // warm L2 two tiles ahead (global_prefetch_b8)
        __builtin_prefetch(zs + (size_t)(pt + 2 * PTILE + sc_r) * D_DIM + sc_cc, 0, 0);
      wait_async<2>();
    } else {
      wait_async<0>();
    }

    if (threadIdx.x < PTILE) {
      ldsZsq[threadIdx.x]   = z_sq[pt + threadIdx.x];
      ldsAlpha[threadIdx.x] = alpha[pt + threadIdx.x];
    }
    __syncthreads();

    // --- 16x16 cross tile: 8 WMMA k-steps over K=256.
    const _Float16* Abuf = &ldsA[i & 1][0];
    v8f cacc = {};
    #pragma unroll
    for (int kk = 0; kk < 8; ++kk) {
      const int kb = kk * 32 + koff;
      const _Float16* ap = Abuf + a_row * LDS_A_STRIDE + kb;
      v8h a0 = *(const v8h*)ap;
      v8h a1 = *(const v8h*)(ap + 16);
      v16h af;
      #pragma unroll
      for (int i2 = 0; i2 < 8; ++i2) { af[i2] = a0[i2]; af[i2 + 8] = a1[i2]; }
      cacc = __builtin_amdgcn_wmma_f32_16x16x32_f16(
          /*neg_a=*/false, af, /*neg_b=*/false, bfrag[kk],
          /*c_mod=*/(short)0, cacc, /*reuse_a=*/false, /*reuse_b=*/false);
    }

    // --- Fused epilogue. C layout: VGPR r, lane L -> M = r + 8*(L>>4), N = L&15.
    #pragma unroll
    for (int r = 0; r < 8; ++r) {
      const int pl  = r + hi * 8;
      const float d = ldsZsq[pl] + my_xsq - 2.0f * cacc[r];
      acc[r] += ldsAlpha[pl] * __expf(-d);
    }
    __syncthreads();
  }

  // --- Reduce over the 8 row-slots, then across the two half-waves
  // (lane L and L+16 cover the same column with disjoint p sets).
  float s = 0.0f;
  #pragma unroll
  for (int r = 0; r < 8; ++r) s += acc[r];
  s += __shfl_xor(s, 16, 32);
  if (hi == 0) out[col] = s;
}

// ---------------------------------------------------------------------------
extern "C" void kernel_launch(void* const* d_in, const int* in_sizes, int n_in,
                              void* d_out, int out_size, void* d_ws, size_t ws_size,
                              hipStream_t stream) {
  const float* x     = (const float*)d_in[0];   // [N, D]
  const float* z     = (const float*)d_in[1];   // [P, D]
  const float* alpha = (const float*)d_in[2];   // [P]
  const float* ls    = (const float*)d_in[3];   // [D]

  const int P = in_sizes[2];
  const int D = in_sizes[3];          // 256
  const int N = in_sizes[0] / D;      // 65536

  // Workspace layout: xs_h [N*D f16] | zs_h [P*D f16] | x_sq [N f32] | z_sq [P f32]
  _Float16* xs_h = (_Float16*)d_ws;
  _Float16* zs_h = xs_h + (size_t)N * D;
  float*    x_sq = (float*)(zs_h + (size_t)P * D);
  float*    z_sq = x_sq + N;

  gpr_prep<<<N, 256, 0, stream>>>(x, ls, xs_h, x_sq, D);
  gpr_prep<<<P, 256, 0, stream>>>(z, ls, zs_h, z_sq, D);

  gpr_main<<<N / NCOLS_PER_BLOCK, 256, 0, stream>>>(
      xs_h, zs_h, x_sq, z_sq, alpha, (float*)d_out, P);
}
